// Quantize_45157286150852
// MI455X (gfx1250) — matrix operand order repaired and checked by reading, not compiled, not run on hardware
//
#include <hip/hip_runtime.h>
#include <hip/hip_bf16.h>

typedef __attribute__((ext_vector_type(16))) _Float16 v16h;
typedef __attribute__((ext_vector_type(8)))  _Float16 v8h;
typedef __attribute__((ext_vector_type(8)))  float    v8f;
typedef __attribute__((ext_vector_type(4)))  unsigned int v4u;
typedef __attribute__((ext_vector_type(8)))  int      v8i;
typedef __attribute__((ext_vector_type(4)))  int      v4i;

#define NTOK   65536          // B*T = 32*2048
#define NDIM   512            // INPUT_DIM
#define KSPEC  544            // padded 2*257 -> 17*32 (interleaved re/im spectrum)
#define DDIM   256            // VQ_DIM
#define NEMB   1024
#define NCB    4
#define TWOPI_OVER_N 0.012271846303085129f  // 2*pi/512

union V16 { v16h v; v8h h[2]; };

__device__ __forceinline__ v8f wmma_f16(v8h alo, v8h ahi, v8h blo, v8h bhi, v8f c) {
  V16 a, b; a.h[0] = alo; a.h[1] = ahi; b.h[0] = blo; b.h[1] = bhi;
  return __builtin_amdgcn_wmma_f32_16x16x32_f16(false, a.v, false, b.v, (short)0, c, false, false);
}

// ---------------------------------------------------------------------------
// Tensor Data Mover: 2D tile load Global -> LDS.
// D# packed per CDNA5 ISA:
//   group0: count=1 | lds_addr | global_addr | type=2
//   group1: data_size, pad_enable/interval/amount, tensor dims/strides, tile dims
// All units in data_size elements (2 bytes here). One descriptor per wave.
// ---------------------------------------------------------------------------
__device__ __forceinline__ void tdm_load_2d(const void* gptr, void* lptr,
    unsigned tensor_d0, unsigned tensor_d1, unsigned stride0,
    unsigned tile_d0, unsigned tile_d1,
    unsigned pad_interval_code, unsigned pad_amount_code) {
  unsigned long long ga = (unsigned long long)(uintptr_t)gptr;
  unsigned lds = (unsigned)(unsigned long long)(uintptr_t)lptr;  // low 32 bits = LDS offset
  v4u g0;
  g0[0] = 1u;                                             // count=1 (valid), load, not restore
  g0[1] = lds;                                            // lds_addr (bytes)
  g0[2] = (unsigned)(ga & 0xffffffffu);                   // global_addr[31:0]
  g0[3] = (unsigned)((ga >> 32) & 0x01ffffffu) | (2u << 30);  // global_addr[56:32] | type=2
  v8i g1;
  g1[0] = (int)((1u << 16)                                // data_size = 1 -> 2 bytes
              | (1u << 20)                                // pad_enable
              | (pad_interval_code << 22)
              | (pad_amount_code << 25));
  g1[1] = (int)((tensor_d0 & 0xffffu) << 16);             // abar_addr=0 | tensor_dim0[15:0]
  g1[2] = (int)((tensor_d0 >> 16) | ((tensor_d1 & 0xffffu) << 16));
  g1[3] = (int)((tensor_d1 >> 16) | (tile_d0 << 16));     // tensor_dim1[31:16] | tile_dim0
  g1[4] = (int)(tile_d1 & 0xffffu);                       // tile_dim1 | tile_dim2=0
  g1[5] = (int)stride0;                                   // tensor_dim0_stride[31:0]
  g1[6] = 0;                                              // stride0[47:32]=0 | stride1 lo=0
  g1[7] = 0;
  v4i g2 = {0, 0, 0, 0};
  v4i g3 = {0, 0, 0, 0};
#if __clang_major__ >= 23
  v8i gz = {0, 0, 0, 0, 0, 0, 0, 0};
  __builtin_amdgcn_tensor_load_to_lds(g0, g1, g2, g3, gz, 0);
#else
  __builtin_amdgcn_tensor_load_to_lds(g0, g1, g2, g3, 0);
#endif
}

// ---------------------------------------------------------------------------
// Kernel A: generate DFT twiddle matrix W (512 x 544) and fold matrix
// CP = irfft-matrix @ projector (544 x 256), both written directly in the
// wave32 WMMA B-fragment layout:  half_index = ((t*KB + kb)*32 + lane)*16 + h
// lanes 0-15 hold K = kb*32 + (h<8 ? h : h+8), lanes 16-31 hold K = kb*32 + 8 + (h<8 ? h : h+8),
// column n = t*16 + (lane & 15).
// Spectrum column layout: col 2k = Re_k, col 2k+1 = Im_k (k = 0..256).
// ---------------------------------------------------------------------------
__global__ void gen_tables_kernel(const float* __restrict__ proj,
                                  _Float16* __restrict__ WbB,
                                  _Float16* __restrict__ cpB) {
  const int WBH = 512 * KSPEC;
  const int CPH = KSPEC * DDIM;
  int idx = blockIdx.x * 256 + threadIdx.x;
  if (idx < WBH) {
    int h    = idx & 15;
    int lane = (idx >> 4) & 31;
    int rest = idx >> 9;
    int kb   = rest & 15;
    int t    = rest >> 4;
    int n    = kb * 32 + ((h >> 3) << 4) + ((lane >> 4) << 3) + (h & 7);
    int col  = t * 16 + (lane & 15);
    float val = 0.f;
    if (col < 514) {
      int kbin = col >> 1;
      float ang = (float)((n * kbin) & 511) * TWOPI_OVER_N;
      float s, c; sincosf(ang, &s, &c);
      val = (col & 1) ? -s : c;
    }
    WbB[idx] = (_Float16)val;
  } else if (idx < WBH + CPH) {
    int j    = idx - WBH;
    int h    = j & 15;
    int lane = (j >> 4) & 31;
    int rest = j >> 9;
    int kb   = rest % 17;
    int t    = rest / 17;
    int row  = kb * 32 + ((h >> 3) << 4) + ((lane >> 4) << 3) + (h & 7);
    int d    = t * 16 + (lane & 15);
    float val = 0.f;
    if (row < 514) {
      int kbin = row >> 1;
      int odd  = row & 1;
      float acc = 0.f;
      for (int n = 0; n < NDIM; ++n) {
        float ang = (float)((n * kbin) & 511) * TWOPI_OVER_N;
        float s, c; sincosf(ang, &s, &c);
        float tg = odd ? -s : c;
        acc = fmaf(tg, proj[n * DDIM + d], acc);
      }
      float w = (kbin == 0 || kbin == 256) ? 1.0f : 2.0f;
      val = acc * (w * (1.0f / 512.0f));
    }
    cpB[j] = (_Float16)val;
  }
}

// ---------------------------------------------------------------------------
// Kernel B: normalize codebook rows, write transposed into WMMA B-fragment
// layout (K dim = d, N dim = m within codebook c).
// ---------------------------------------------------------------------------
__global__ void codebook_kernel(const float* __restrict__ cb,
                                _Float16* __restrict__ cbB) {
  int row  = blockIdx.x;            // c*1024 + m
  int d    = threadIdx.x;           // 0..255
  int lane = d & 31, wid = d >> 5;
  float v  = cb[(size_t)row * DDIM + d];
  float ss = v * v;
  #pragma unroll
  for (int m = 16; m >= 1; m >>= 1) ss += __shfl_xor(ss, m, 32);
  __shared__ float sred[8];
  if (lane == 0) sred[wid] = ss;
  __syncthreads();
  float tot = 0.f;
  #pragma unroll
  for (int i = 0; i < 8; ++i) tot += sred[i];
  float nv = v * rsqrtf(tot);
  int c = row >> 10, m = row & 1023;
  int t = m >> 4;
  int kb = d >> 5, rem = d & 31;
  int blane = ((rem >> 3) & 1) * 16 + (m & 15);
  int h = (rem & 7) + ((rem >> 4) << 3);
  size_t off = ((((size_t)c * 64 + t) * 8 + kb) * 32 + blane) * 16 + h;
  cbB[off] = (_Float16)nv;
}

// ---------------------------------------------------------------------------
// Kernel C: theta1 per token: X1 = sum x_n e^{-2pi i n/512}; theta = atan2(Im,Re)
// ---------------------------------------------------------------------------
__global__ void theta_kernel(const float* __restrict__ x, float* __restrict__ theta) {
  __shared__ float ct[512], st[512];
  int tid = threadIdx.x;
  for (int j = tid; j < 512; j += 256) {
    float ang = (float)j * TWOPI_OVER_N;
    sincosf(ang, &st[j], &ct[j]);
  }
  __syncthreads();
  int tok = blockIdx.x * 256 + tid;
  const float* xr = x + (size_t)tok * NDIM;
  float re = 0.f, im = 0.f;
  for (int n = 0; n < NDIM; ++n) {
    float xv = xr[n];
    re = fmaf(xv,  ct[n], re);
    im = fmaf(xv, -st[n], im);
  }
  theta[tok] = atan2f(im, re);
}

// ---------------------------------------------------------------------------
// Kernel C2: x f32 -> f16 (one pass, so the GEMM tiles can be TDM-DMA'd)
// ---------------------------------------------------------------------------
__global__ void convert_kernel(const float* __restrict__ x, _Float16* __restrict__ xh) {
  size_t i = ((size_t)blockIdx.x * 256 + threadIdx.x) * 8;
  float4 a = *(const float4*)(x + i);
  float4 b = *(const float4*)(x + i + 4);
  v8h o;
  o[0] = (_Float16)a.x; o[1] = (_Float16)a.y; o[2] = (_Float16)a.z; o[3] = (_Float16)a.w;
  o[4] = (_Float16)b.x; o[5] = (_Float16)b.y; o[6] = (_Float16)b.z; o[7] = (_Float16)b.w;
  *(v8h*)(xh + i) = o;
}

// ---------------------------------------------------------------------------
// Kernel D: spec = (x @ W) rotated by e^{-ik*theta1}, stored f16.
// 8 waves (4M x 2N), block tile 64 rows x 32 cols. Each wave issues ONE TDM
// descriptor DMA-ing its 64x64-half x-slice into a padded (stride-72) LDS
// slice; pad_interval=32 DW, pad_amount=4 DW. After s_wait_tensorcnt + one
// barrier the 16-step K loop is pure ds_read_b128 + global_load_b128 + WMMA,
// with the phase rotation (lane-pair shfl_xor) fused in the epilogue.
// ---------------------------------------------------------------------------
__global__ void __launch_bounds__(256) dft_rot_kernel(const _Float16* __restrict__ xh,
                                                      const _Float16* __restrict__ WbB,
                                                      const float* __restrict__ theta,
                                                      _Float16* __restrict__ spec) {
  __shared__ __align__(16) _Float16 at[8][64 * 72];   // 8 slices, padded stride 72
  __shared__ float th[64];
  int tid = threadIdx.x;
  int lane = tid & 31, wid = tid >> 5;
  int waveM = wid >> 1, waveN = wid & 1;
  int rowbase = blockIdx.x * 64;
  int tglob = blockIdx.y * 2 + waveN;                 // 0..33
  int swid = __builtin_amdgcn_readfirstlane(wid);     // wave-uniform in SGPR
  // slice swid: x columns [swid*64, swid*64+64) for 64 rows
  tdm_load_2d(xh + (size_t)rowbase * NDIM + swid * 64, &at[swid][0],
              /*tensor_d0=*/NDIM, /*tensor_d1=*/NTOK, /*stride0=*/NDIM,
              /*tile_d0=*/64, /*tile_d1=*/64,
              /*pad_interval(32DW)=*/4, /*pad_amount(4DW)=*/3);
  if (tid < 64) th[tid] = theta[rowbase + tid];
  __builtin_amdgcn_s_wait_tensorcnt(0);
  __syncthreads();

  v8f acc = {};
  int k0 = (lane >> 4) * 8;
  int frow = waveM * 16 + (lane & 15);
  for (int kb = 0; kb < 16; ++kb) {
    int s  = kb >> 1;
    int co = (kb & 1) * 32;
    v8h alo = *(const v8h*)&at[s][frow * 72 + co + k0];
    v8h ahi = *(const v8h*)&at[s][frow * 72 + co + k0 + 16];
    const _Float16* bp = WbB + ((size_t)(tglob * 16 + kb) * 32 + lane) * 16;
    v8h blo = *(const v8h*)bp;
    v8h bhi = *(const v8h*)(bp + 8);
    acc = wmma_f16(alo, ahi, blo, bhi, acc);
  }
  int gcol = tglob * 16 + (lane & 15);
  int kbin = gcol >> 1;
  int odd  = gcol & 1;
  #pragma unroll
  for (int r = 0; r < 8; ++r) {
    int rloc = waveM * 16 + r + ((lane >> 4) << 3);
    float v = acc[r];
    float p = __shfl_xor(v, 1, 32);   // partner (im for re lanes, re for im lanes)
    float a = (float)kbin * th[rloc];
    float s, c; sincosf(a, &s, &c);
    float nv = odd ? (v * c - p * s) : (v * c + p * s);
    spec[(size_t)(rowbase + rloc) * KSPEC + gcol] = (_Float16)nv;
  }
}

// ---------------------------------------------------------------------------
// Kernel E: feat = spec @ CP, fused L2 row-normalization, store f16.
// 8 waves 2(M) x 4(N); tile 32 rows x 256 cols (full feature row).
// ---------------------------------------------------------------------------
__global__ void __launch_bounds__(256) proj_norm_kernel(const _Float16* __restrict__ spec,
                                                        const _Float16* __restrict__ cpB,
                                                        _Float16* __restrict__ featn) {
  __shared__ __align__(16) _Float16 at[32 * 40];
  __shared__ float ssArr[32];
  int tid = threadIdx.x;
  int lane = tid & 31, wid = tid >> 5;
  int waveM = wid >> 2, waveN = wid & 3;
  int rowbase = blockIdx.x * 32;
  v8f acc[4] = {};
  int arow = tid >> 3;
  int acol = (tid & 7) * 4;
  int k0 = (lane >> 4) * 8;
  int frow = waveM * 16 + (lane & 15);
  for (int kb = 0; kb < 17; ++kb) {
    const _Float16* src = spec + (size_t)(rowbase + arow) * KSPEC + kb * 32 + acol;
    *(uint2*)&at[arow * 40 + acol] = *(const uint2*)src;
    __syncthreads();
    v8h alo = *(const v8h*)&at[frow * 40 + k0];
    v8h ahi = *(const v8h*)&at[frow * 40 + k0 + 16];
    #pragma unroll
    for (int nt = 0; nt < 4; ++nt) {
      int t = waveN * 4 + nt;
      const _Float16* bp = cpB + ((size_t)(t * 17 + kb) * 32 + lane) * 16;
      v8h blo = *(const v8h*)bp;
      v8h bhi = *(const v8h*)(bp + 8);
      acc[nt] = wmma_f16(alo, ahi, blo, bhi, acc[nt]);
    }
    __syncthreads();
  }
  if (tid < 32) ssArr[tid] = 0.f;
  __syncthreads();
  #pragma unroll
  for (int r = 0; r < 8; ++r) {
    float p = 0.f;
    #pragma unroll
    for (int nt = 0; nt < 4; ++nt) { float v = acc[nt][r]; p = fmaf(v, v, p); }
    #pragma unroll
    for (int m = 8; m >= 1; m >>= 1) p += __shfl_xor(p, m, 32);
    if ((lane & 15) == 0) {
      int rloc = waveM * 16 + r + ((lane >> 4) << 3);
      atomicAdd(&ssArr[rloc], p);       // ds_add_f32 across the 4 N-waves
    }
  }
  __syncthreads();
  #pragma unroll
  for (int r = 0; r < 8; ++r) {
    int rloc = waveM * 16 + r + ((lane >> 4) << 3);
    float rsq = rsqrtf(ssArr[rloc]);
    #pragma unroll
    for (int nt = 0; nt < 4; ++nt) {
      int col = waveN * 64 + nt * 16 + (lane & 15);
      featn[(size_t)(rowbase + rloc) * DDIM + col] = (_Float16)(acc[nt][r] * rsq);
    }
  }
}

// ---------------------------------------------------------------------------
// Kernel F: similarity GEMM + fused argmax. Block = 32 rows x one codebook
// (1024 cols). The 32x256 featn tile is DMA'd by TDM: one descriptor per wave
// (4 rows each), pad_interval=128 DW + pad_amount=4 DW -> stride-264 LDS
// layout (bank-conflict free for ds_read_b128 fragments). Running (max,idx)
// per row; no similarity tensor ever touches HBM.
// ---------------------------------------------------------------------------
__global__ void __launch_bounds__(256) argmax_kernel(const _Float16* __restrict__ featn,
                                                     const _Float16* __restrict__ cbB,
                                                     int* __restrict__ out) {
  __shared__ __align__(16) _Float16 at[32 * 264];
  __shared__ float bvArr[32][4];
  __shared__ int   biArr[32][4];
  int tid = threadIdx.x;
  int lane = tid & 31, wid = tid >> 5;
  int waveM = wid >> 2, waveN = wid & 3;
  int rowbase = blockIdx.x * 32;
  int c = blockIdx.y;
  int swid = __builtin_amdgcn_readfirstlane(wid);
  tdm_load_2d(featn + (size_t)(rowbase + swid * 4) * DDIM, &at[swid * 4 * 264],
              /*tensor_d0=*/DDIM, /*tensor_d1=*/NTOK, /*stride0=*/DDIM,
              /*tile_d0=*/DDIM, /*tile_d1=*/4,
              /*pad_interval(128DW)=*/6, /*pad_amount(4DW)=*/3);
  __builtin_amdgcn_s_wait_tensorcnt(0);
  __syncthreads();

  int k0 = (lane >> 4) * 8;
  int frow = waveM * 16 + (lane & 15);
  float bestv[8]; int besti[8];
  #pragma unroll
  for (int r = 0; r < 8; ++r) { bestv[r] = -3.0e38f; besti[r] = 0; }
  for (int nt = 0; nt < 16; ++nt) {
    v8f acc = {};
    int t = waveN * 16 + nt;
    #pragma unroll
    for (int kb = 0; kb < 8; ++kb) {
      v8h alo = *(const v8h*)&at[frow * 264 + kb * 32 + k0];
      v8h ahi = *(const v8h*)&at[frow * 264 + kb * 32 + k0 + 16];
      const _Float16* bp = cbB + ((((size_t)c * 64 + t) * 8 + kb) * 32 + lane) * 16;
      v8h blo = *(const v8h*)bp;
      v8h bhi = *(const v8h*)(bp + 8);
      acc = wmma_f16(alo, ahi, blo, bhi, acc);
    }
    int gcol = waveN * 256 + nt * 16 + (lane & 15);
    #pragma unroll
    for (int r = 0; r < 8; ++r) {
      float v = acc[r];
      if (v > bestv[r]) { bestv[r] = v; besti[r] = gcol; }
    }
  }
  #pragma unroll
  for (int r = 0; r < 8; ++r) {
    #pragma unroll
    for (int m = 1; m < 16; m <<= 1) {
      float ov = __shfl_xor(bestv[r], m, 32);
      int   oi = __shfl_xor(besti[r], m, 32);
      if (ov > bestv[r] || (ov == bestv[r] && oi < besti[r])) { bestv[r] = ov; besti[r] = oi; }
    }
  }
  if ((lane & 15) == 0) {
    #pragma unroll
    for (int r = 0; r < 8; ++r) {
      int rloc = waveM * 16 + r + ((lane >> 4) << 3);
      bvArr[rloc][waveN] = bestv[r];
      biArr[rloc][waveN] = besti[r];
    }
  }
  __syncthreads();
  if (tid < 32) {
    float bv = bvArr[tid][0]; int bi = biArr[tid][0];
    #pragma unroll
    for (int w = 1; w < 4; ++w) {
      float ov = bvArr[tid][w]; int oi = biArr[tid][w];
      if (ov > bv || (ov == bv && oi < bi)) { bv = ov; bi = oi; }
    }
    out[(size_t)(rowbase + tid) * NCB + c] = bi;
  }
}

// ---------------------------------------------------------------------------
extern "C" void kernel_launch(void* const* d_in, const int* in_sizes, int n_in,
                              void* d_out, int out_size, void* d_ws, size_t ws_size,
                              hipStream_t stream) {
  const float* x    = (const float*)d_in[0];   // (32,2048,512)
  const float* proj = (const float*)d_in[1];   // (512,256)
  const float* cb   = (const float*)d_in[2];   // (4,1024,256)
  int* out = (int*)d_out;

  char* ws = (char*)d_ws;
  size_t off = 0;
  _Float16* spec  = (_Float16*)(ws + off); off += (size_t)NTOK * KSPEC * 2;
  _Float16* featn = (_Float16*)(ws + off); off += (size_t)NTOK * DDIM * 2;
  _Float16* xh    = (_Float16*)(ws + off); off += (size_t)NTOK * NDIM * 2;
  _Float16* WbB   = (_Float16*)(ws + off); off += (size_t)512 * KSPEC * 2;
  _Float16* cpB   = (_Float16*)(ws + off); off += (size_t)KSPEC * DDIM * 2;
  _Float16* cbB   = (_Float16*)(ws + off); off += (size_t)NCB * NEMB * DDIM * 2;
  float*    theta = (float*)(ws + off);    off += (size_t)NTOK * 4;

  const int totalGen = (512 * KSPEC + KSPEC * DDIM);   // 417792 halves
  gen_tables_kernel<<<(totalGen + 255) / 256, 256, 0, stream>>>(proj, WbB, cpB);
  codebook_kernel<<<NCB * NEMB, 256, 0, stream>>>(cb, cbB);
  theta_kernel<<<NTOK / 256, 256, 0, stream>>>(x, theta);
  convert_kernel<<<(NTOK * NDIM / 8) / 256, 256, 0, stream>>>(x, xh);
  dft_rot_kernel<<<dim3(NTOK / 64, KSPEC / 32), 256, 0, stream>>>(xh, WbB, theta, spec);
  proj_norm_kernel<<<NTOK / 32, 256, 0, stream>>>(spec, cpB, featn);
  argmax_kernel<<<dim3(NTOK / 32, NCB), 256, 0, stream>>>(featn, cbB, out);
}
